// RiemannianClassifier_37495064494758
// MI455X (gfx1250) — matrix operand order repaired and checked by reading, not compiled, not run on hardware
//
#include <hip/hip_runtime.h>
#include <hip/hip_bf16.h>

typedef __attribute__((ext_vector_type(16))) __bf16 v16bf;
typedef __attribute__((ext_vector_type(8)))  __bf16 v8bf;
typedef __attribute__((ext_vector_type(4)))  __bf16 v4bf;
typedef __attribute__((ext_vector_type(8)))  float  v8f;
typedef __attribute__((ext_vector_type(4)))  unsigned int u32x4;
typedef __attribute__((ext_vector_type(8)))  int i32x8;
typedef __attribute__((ext_vector_type(4)))  int i32x4;

#define DK    512          // K dimension (feature dim)
#define BM    128          // block tile M
#define BN    64           // block tile N
#define BK    32           // K step = one bf16 WMMA K depth
#define LDT   40           // padded LDS row stride (legacy path)

#if defined(__has_builtin)
#  if __has_builtin(__builtin_amdgcn_tensor_load_to_lds)
#    define HAVE_TDM 1
#  endif
#endif
#ifndef HAVE_TDM
#  define HAVE_TDM 0
#endif

union FragU { v16bf v; v8bf h8[2]; };
union Pk2   { __bf16 b[2]; unsigned u; };

// ---------------------------------------------------------------------------
// Pass 1a: norms only (legacy fallback path)
// ---------------------------------------------------------------------------
__global__ __launch_bounds__(256)
void rc_norms(const float* __restrict__ src, float* __restrict__ dst) {
    const int row = blockIdx.x;
    const float* p = src + (size_t)row * DK;
    float s = 0.0f;
    for (int j = threadIdx.x; j < DK; j += 256) {
        float v = (j == 0) ? 0.0f : p[j];
        s += v * v;
    }
    __shared__ float red[256];
    red[threadIdx.x] = s;
    __syncthreads();
    for (int off = 128; off > 0; off >>= 1) {
        if (threadIdx.x < off) red[threadIdx.x] += red[threadIdx.x + off];
        __syncthreads();
    }
    if (threadIdx.x == 0) dst[row] = sqrtf(1.0f + red[0]);
}

// ---------------------------------------------------------------------------
// Pass 1b (TDM path): fused norm + fp32 -> bf16 hi/lo pack, time component
// folded into column 0 (sign = -1 for A, +1 for B).
// ---------------------------------------------------------------------------
__global__ __launch_bounds__(256)
void rc_pack(const float* __restrict__ src, __bf16* __restrict__ hiP,
             __bf16* __restrict__ loP, float sign) {
    const int row = blockIdx.x;
    const int t   = threadIdx.x;
    float2 q = *(const float2*)(src + (size_t)row * DK + 2 * t);
    float a0 = (t == 0) ? 0.0f : q.x;
    float s  = a0 * a0 + q.y * q.y;
    __shared__ float red[256];
    red[t] = s;
    __syncthreads();
    for (int off = 128; off > 0; off >>= 1) {
        if (t < off) red[t] += red[t + off];
        __syncthreads();
    }
    if (t == 0) q.x = sign * sqrtf(1.0f + red[0]);
    Pk2 ph, pl;
    __bf16 h0 = (__bf16)q.x; ph.b[0] = h0; pl.b[0] = (__bf16)(q.x - (float)h0);
    __bf16 h1 = (__bf16)q.y; ph.b[1] = h1; pl.b[1] = (__bf16)(q.y - (float)h1);
    ((unsigned*)hiP)[(size_t)row * (DK / 2) + t] = ph.u;
    ((unsigned*)loP)[(size_t)row * (DK / 2) + t] = pl.u;
}

// ---------------------------------------------------------------------------
// TDM descriptor build + issue: 2D tile (tile_d0 x tile_d1) of bf16,
// tensor row length 512, row stride 512, into LDS at lds_off.
// ---------------------------------------------------------------------------
#if HAVE_TDM
__device__ __forceinline__ void tdm_load_2d(unsigned lds_off, const __bf16* g,
                                            unsigned tile_d0, unsigned tile_d1,
                                            unsigned tensor_d1) {
    unsigned long long ga = (unsigned long long)(uintptr_t)g;
    u32x4 g0;
    g0[0] = 1u;                                    // count=1, user descriptor
    g0[1] = lds_off;                               // LDS byte address
    g0[2] = (unsigned)ga;                          // global_addr[31:0]
    g0[3] = (unsigned)((ga >> 32) & 0x01ffffffu)   // global_addr[56:32]
            | 0x80000000u;                         // type = 2 ("image")
    i32x8 g1;
    g1[0] = (int)(1u << 16);                       // data_size = 1 (2 bytes)
    g1[1] = (int)((DK & 0xffffu) << 16);           // tensor_dim0[15:0]
    g1[2] = (int)((DK >> 16) | ((tensor_d1 & 0xffffu) << 16)); // dim0 hi | dim1 lo
    g1[3] = (int)((tensor_d1 >> 16) | (tile_d0 << 16));        // dim1 hi | tile_dim0
    g1[4] = (int)tile_d1;                          // tile_dim1 (tile_dim2 = 0)
    g1[5] = (int)DK;                               // tensor_dim0_stride
    g1[6] = 0;
    g1[7] = 0;
    i32x4 z4 = {0, 0, 0, 0};
#if __clang_major__ >= 23
    i32x8 z8 = {0, 0, 0, 0, 0, 0, 0, 0};
    __builtin_amdgcn_tensor_load_to_lds(g0, g1, z4, z4, z8, 0);
#else
    __builtin_amdgcn_tensor_load_to_lds(g0, g1, z4, z4, 0);
#endif
}
#endif

// ---------------------------------------------------------------------------
// Pass 2 (TDM path): bf16x3 WMMA GEMM, tiles DMA'd by the Tensor Data Mover
// into double-buffered LDS; TENSORcnt + workgroup barrier pipeline.
// ---------------------------------------------------------------------------
__global__ __launch_bounds__(256)
void rc_gemm_tdm(const __bf16* __restrict__ Ahi, const __bf16* __restrict__ Alo,
                 const __bf16* __restrict__ Bhi, const __bf16* __restrict__ Blo,
                 float* __restrict__ out, int Mrows, int Ncols) {
    __shared__ __bf16 sA[2][2][BM * BK];   // [buf][hi/lo]
    __shared__ __bf16 sB[2][2][BN * BK];

    const int tid  = threadIdx.x;
    const int wave = tid >> 5;
    const int lane = tid & 31;
    const int waveM = wave >> 1;
    const int waveN = wave & 1;
    const int h   = lane >> 4;
    const int l16 = lane & 15;

    const int rowBase = blockIdx.y * BM;
    const int colBase = blockIdx.x * BN;

    v8f acc[2][2];
#pragma unroll
    for (int a = 0; a < 2; ++a)
#pragma unroll
        for (int b = 0; b < 2; ++b)
#pragma unroll
            for (int r = 0; r < 8; ++r) acc[a][b][r] = 0.0f;

#if HAVE_TDM
    // ---- prologue: DMA first K tile into buffer 0 ----
    if (wave == 0) {
        tdm_load_2d((unsigned)(uintptr_t)&sA[0][0][0], Ahi + (size_t)rowBase * DK, BK, BM, (unsigned)Mrows);
        tdm_load_2d((unsigned)(uintptr_t)&sA[0][1][0], Alo + (size_t)rowBase * DK, BK, BM, (unsigned)Mrows);
        tdm_load_2d((unsigned)(uintptr_t)&sB[0][0][0], Bhi + (size_t)colBase * DK, BK, BN, (unsigned)Ncols);
        tdm_load_2d((unsigned)(uintptr_t)&sB[0][1][0], Blo + (size_t)colBase * DK, BK, BN, (unsigned)Ncols);
    }
#endif

    for (int kt = 0; kt < DK / BK; ++kt) {
        const int buf = kt & 1;
#if HAVE_TDM
        if (wave == 0) __builtin_amdgcn_s_wait_tensorcnt(0);
        __syncthreads();                       // tile[kt] visible to all waves
        if (wave == 0 && kt + 1 < DK / BK) {   // DMA tile[kt+1] into other buf
            const int k1 = (kt + 1) * BK;
            const int nb = buf ^ 1;
            tdm_load_2d((unsigned)(uintptr_t)&sA[nb][0][0], Ahi + (size_t)rowBase * DK + k1, BK, BM, (unsigned)Mrows);
            tdm_load_2d((unsigned)(uintptr_t)&sA[nb][1][0], Alo + (size_t)rowBase * DK + k1, BK, BM, (unsigned)Mrows);
            tdm_load_2d((unsigned)(uintptr_t)&sB[nb][0][0], Bhi + (size_t)colBase * DK + k1, BK, BN, (unsigned)Ncols);
            tdm_load_2d((unsigned)(uintptr_t)&sB[nb][1][0], Blo + (size_t)colBase * DK + k1, BK, BN, (unsigned)Ncols);
        }
#else
        // fallback: cooperative copy of packed bf16 tiles
        __syncthreads();
        const int k0 = kt * BK;
        for (int i = tid; i < BM * BK / 8; i += 256) {
            const int r = i >> 2, c8 = (i & 3) * 8;
            *(uint4*)&sA[buf][0][r * BK + c8] = *(const uint4*)(Ahi + (size_t)(rowBase + r) * DK + k0 + c8);
            *(uint4*)&sA[buf][1][r * BK + c8] = *(const uint4*)(Alo + (size_t)(rowBase + r) * DK + k0 + c8);
        }
        for (int i = tid; i < BN * BK / 8; i += 256) {
            const int r = i >> 2, c8 = (i & 3) * 8;
            *(uint4*)&sB[buf][0][r * BK + c8] = *(const uint4*)(Bhi + (size_t)(colBase + r) * DK + k0 + c8);
            *(uint4*)&sB[buf][1][r * BK + c8] = *(const uint4*)(Blo + (size_t)(colBase + r) * DK + k0 + c8);
        }
        __syncthreads();
#endif
        // ---- fragments per documented wave32 VGPR striping ----
        FragU ahi[2], alo[2], bhi[2], blo[2];
#pragma unroll
        for (int tm = 0; tm < 2; ++tm) {
            const int m = waveM * 32 + tm * 16 + l16;
            ahi[tm].h8[0] = *(const v8bf*)(&sA[buf][0][m * BK + 8 * h]);
            ahi[tm].h8[1] = *(const v8bf*)(&sA[buf][0][m * BK + 16 + 8 * h]);
            alo[tm].h8[0] = *(const v8bf*)(&sA[buf][1][m * BK + 8 * h]);
            alo[tm].h8[1] = *(const v8bf*)(&sA[buf][1][m * BK + 16 + 8 * h]);
        }
#pragma unroll
        for (int tn = 0; tn < 2; ++tn) {
            const int n = waveN * 32 + tn * 16 + l16;
            bhi[tn].h8[0] = *(const v8bf*)(&sB[buf][0][n * BK + 8 * h]);
            bhi[tn].h8[1] = *(const v8bf*)(&sB[buf][0][n * BK + 16 + 8 * h]);
            blo[tn].h8[0] = *(const v8bf*)(&sB[buf][1][n * BK + 8 * h]);
            blo[tn].h8[1] = *(const v8bf*)(&sB[buf][1][n * BK + 16 + 8 * h]);
        }
        // ---- bf16x3: hi*hi + hi*lo + lo*hi ----
#pragma unroll
        for (int tm = 0; tm < 2; ++tm)
#pragma unroll
            for (int tn = 0; tn < 2; ++tn) {
                acc[tm][tn] = __builtin_amdgcn_wmma_f32_16x16x32_bf16(
                    false, ahi[tm].v, false, bhi[tn].v, (short)0, acc[tm][tn], false, false);
                acc[tm][tn] = __builtin_amdgcn_wmma_f32_16x16x32_bf16(
                    false, ahi[tm].v, false, blo[tn].v, (short)0, acc[tm][tn], false, false);
                acc[tm][tn] = __builtin_amdgcn_wmma_f32_16x16x32_bf16(
                    false, alo[tm].v, false, bhi[tn].v, (short)0, acc[tm][tn], false, false);
            }
    }

    const float one_eps = 1.0f + 1e-7f;
#pragma unroll
    for (int tm = 0; tm < 2; ++tm)
#pragma unroll
        for (int tn = 0; tn < 2; ++tn)
#pragma unroll
            for (int r = 0; r < 8; ++r) {
                const int Row = rowBase + waveM * 32 + tm * 16 + 8 * h + r;
                const int Col = colBase + waveN * 32 + tn * 16 + l16;
                const float inner = acc[tm][tn][r];
                const float z = fmaxf(-inner, one_eps);
                const float d = logf(z + sqrtf(z * z - 1.0f));
                out[(size_t)Row * Ncols + Col] = -d;
            }
}

// ---------------------------------------------------------------------------
// Legacy fallback GEMM (fp32 tiles, in-loop bf16 split) — round-1 kernel.
// ---------------------------------------------------------------------------
__global__ __launch_bounds__(256)
void rc_gemm(const float* __restrict__ X, const float* __restrict__ P,
             const float* __restrict__ x0, const float* __restrict__ p0,
             float* __restrict__ out, int Ncols) {
    __shared__ __bf16 sAhi[BM * LDT];
    __shared__ __bf16 sAlo[BM * LDT];
    __shared__ __bf16 sBhi[BN * LDT];
    __shared__ __bf16 sBlo[BN * LDT];

    const int tid  = threadIdx.x;
    const int wave = tid >> 5;
    const int lane = tid & 31;
    const int waveM = wave >> 1;
    const int waveN = wave & 1;
    const int h   = lane >> 4;
    const int l16 = lane & 15;

    const int rowBase = blockIdx.y * BM;
    const int colBase = blockIdx.x * BN;

    v8f acc[2][2];
#pragma unroll
    for (int a = 0; a < 2; ++a)
#pragma unroll
        for (int b = 0; b < 2; ++b)
#pragma unroll
            for (int r = 0; r < 8; ++r) acc[a][b][r] = 0.0f;

    const int ldr = tid >> 3;
    const int ldc = (tid & 7) * 4;

    for (int k0 = 0; k0 < DK; k0 += BK) {
        if (k0 + BK < DK) {
            __builtin_prefetch(X + (size_t)(rowBase + ldr) * DK + (k0 + BK) + ldc, 0, 1);
            __builtin_prefetch(P + (size_t)(colBase + ldr) * DK + (k0 + BK) + ldc, 0, 1);
        }
#pragma unroll
        for (int ps = 0; ps < 4; ++ps) {
            const int r = ldr + ps * 32;
            float4 q = *(const float4*)(X + (size_t)(rowBase + r) * DK + k0 + ldc);
            float f0 = q.x, f1 = q.y, f2 = q.z, f3 = q.w;
            if (k0 == 0 && ldc == 0) f0 = -x0[rowBase + r];
            v4bf vh, vl;
            __bf16 h0 = (__bf16)f0; vh[0] = h0; vl[0] = (__bf16)(f0 - (float)h0);
            __bf16 h1 = (__bf16)f1; vh[1] = h1; vl[1] = (__bf16)(f1 - (float)h1);
            __bf16 h2 = (__bf16)f2; vh[2] = h2; vl[2] = (__bf16)(f2 - (float)h2);
            __bf16 h3 = (__bf16)f3; vh[3] = h3; vl[3] = (__bf16)(f3 - (float)h3);
            *(v4bf*)(&sAhi[r * LDT + ldc]) = vh;
            *(v4bf*)(&sAlo[r * LDT + ldc]) = vl;
        }
#pragma unroll
        for (int ps = 0; ps < 2; ++ps) {
            const int r = ldr + ps * 32;
            float4 q = *(const float4*)(P + (size_t)(colBase + r) * DK + k0 + ldc);
            float f0 = q.x, f1 = q.y, f2 = q.z, f3 = q.w;
            if (k0 == 0 && ldc == 0) f0 = p0[colBase + r];
            v4bf vh, vl;
            __bf16 h0 = (__bf16)f0; vh[0] = h0; vl[0] = (__bf16)(f0 - (float)h0);
            __bf16 h1 = (__bf16)f1; vh[1] = h1; vl[1] = (__bf16)(f1 - (float)h1);
            __bf16 h2 = (__bf16)f2; vh[2] = h2; vl[2] = (__bf16)(f2 - (float)h2);
            __bf16 h3 = (__bf16)f3; vh[3] = h3; vl[3] = (__bf16)(f3 - (float)h3);
            *(v4bf*)(&sBhi[r * LDT + ldc]) = vh;
            *(v4bf*)(&sBlo[r * LDT + ldc]) = vl;
        }
        __syncthreads();

        FragU ahi[2], alo[2], bhi[2], blo[2];
#pragma unroll
        for (int tm = 0; tm < 2; ++tm) {
            const int m = waveM * 32 + tm * 16 + l16;
            ahi[tm].h8[0] = *(const v8bf*)(&sAhi[m * LDT + 8 * h]);
            ahi[tm].h8[1] = *(const v8bf*)(&sAhi[m * LDT + 16 + 8 * h]);
            alo[tm].h8[0] = *(const v8bf*)(&sAlo[m * LDT + 8 * h]);
            alo[tm].h8[1] = *(const v8bf*)(&sAlo[m * LDT + 16 + 8 * h]);
        }
#pragma unroll
        for (int tn = 0; tn < 2; ++tn) {
            const int n = waveN * 32 + tn * 16 + l16;
            bhi[tn].h8[0] = *(const v8bf*)(&sBhi[n * LDT + 8 * h]);
            bhi[tn].h8[1] = *(const v8bf*)(&sBhi[n * LDT + 16 + 8 * h]);
            blo[tn].h8[0] = *(const v8bf*)(&sBlo[n * LDT + 8 * h]);
            blo[tn].h8[1] = *(const v8bf*)(&sBlo[n * LDT + 16 + 8 * h]);
        }
#pragma unroll
        for (int tm = 0; tm < 2; ++tm)
#pragma unroll
            for (int tn = 0; tn < 2; ++tn) {
                acc[tm][tn] = __builtin_amdgcn_wmma_f32_16x16x32_bf16(
                    false, ahi[tm].v, false, bhi[tn].v, (short)0, acc[tm][tn], false, false);
                acc[tm][tn] = __builtin_amdgcn_wmma_f32_16x16x32_bf16(
                    false, ahi[tm].v, false, blo[tn].v, (short)0, acc[tm][tn], false, false);
                acc[tm][tn] = __builtin_amdgcn_wmma_f32_16x16x32_bf16(
                    false, alo[tm].v, false, bhi[tn].v, (short)0, acc[tm][tn], false, false);
            }
        __syncthreads();
    }

    const float one_eps = 1.0f + 1e-7f;
#pragma unroll
    for (int tm = 0; tm < 2; ++tm)
#pragma unroll
        for (int tn = 0; tn < 2; ++tn)
#pragma unroll
            for (int r = 0; r < 8; ++r) {
                const int Row = rowBase + waveM * 32 + tm * 16 + 8 * h + r;
                const int Col = colBase + waveN * 32 + tn * 16 + l16;
                const float inner = acc[tm][tn][r];
                const float z = fmaxf(-inner, one_eps);
                const float d = logf(z + sqrtf(z * z - 1.0f));
                out[(size_t)Row * Ncols + Col] = -d;
            }
}

// ---------------------------------------------------------------------------
extern "C" void kernel_launch(void* const* d_in, const int* in_sizes, int n_in,
                              void* d_out, int out_size, void* d_ws, size_t ws_size,
                              hipStream_t stream) {
    const float* x   = (const float*)d_in[0];   // (B, 512) fp32
    const float* pts = (const float*)d_in[1];   // (C, 512) fp32
    const int B = in_sizes[0] / DK;             // 16384
    const int C = in_sizes[1] / DK;             // 2048

    const size_t planeA = (size_t)B * DK * sizeof(__bf16);
    const size_t planeB = (size_t)C * DK * sizeof(__bf16);
    const size_t needPacked = 2 * planeA + 2 * planeB;

    dim3 grid(C / BN, B / BM);                  // (32, 128)

    if (ws_size >= needPacked) {
        // --- packed bf16 + Tensor Data Mover path ---
        __bf16* Ahi = (__bf16*)d_ws;
        __bf16* Alo = (__bf16*)((char*)d_ws + planeA);
        __bf16* Bhi = (__bf16*)((char*)d_ws + 2 * planeA);
        __bf16* Blo = (__bf16*)((char*)d_ws + 2 * planeA + planeB);
        rc_pack<<<B, 256, 0, stream>>>(x,   Ahi, Alo, -1.0f);
        rc_pack<<<C, 256, 0, stream>>>(pts, Bhi, Blo, +1.0f);
        rc_gemm_tdm<<<grid, 256, 0, stream>>>(Ahi, Alo, Bhi, Blo,
                                              (float*)d_out, B, C);
    } else {
        // --- fallback: fp32 staging with in-loop split ---
        float* x0 = (float*)d_ws;
        float* p0 = x0 + B;
        rc_norms<<<B, 256, 0, stream>>>(x,   x0);
        rc_norms<<<C, 256, 0, stream>>>(pts, p0);
        rc_gemm<<<grid, 256, 0, stream>>>(x, pts, x0, p0, (float*)d_out, C);
    }
}
